// KVCacheCompressor_75376676045060
// MI455X (gfx1250) — matrix operand order repaired and checked by reading, not compiled, not run on hardware
//
#include <hip/hip_runtime.h>
#include <hip/hip_bf16.h>

// CDNA5 / gfx1250 WMMA types (wave32)
typedef __attribute__((ext_vector_type(16))) __bf16 v16bf;
typedef __attribute__((ext_vector_type(8)))  float  v8f;

#define DIM      128   // feature dim
#define NTILES   16    // 256 clusters / 16 cols per WMMA tile
#define KCHUNKS  4     // 128 / 32 k-steps per WMMA
#define BPACK_N  (KCHUNKS * NTILES * 32 * 16)   // 32768 bf16 = 64 KB

// ---------------------------------------------------------------------------
// Zero the accumulator region of the workspace.
// ---------------------------------------------------------------------------
__global__ void kmeans_zero(float* __restrict__ ws, int n) {
    int i = blockIdx.x * blockDim.x + threadIdx.x;
    if (i < n) ws[i] = 0.0f;
}

// ---------------------------------------------------------------------------
// Pack key_centroids [K=256, D=128] fp32 into bf16 WMMA B-fragments and
// compute per-centroid squared norms.
//
// B (32x16 bf16) fragment, contiguous k order used here (same as A): lanes 0-15 hold N=lane,
// K=0..15 (2 per VGPR); lanes 16-31 hold N=lane-16, K=16..31.
// Linear pack index: (((kchunk*NTILES + tile)*32 + lane)*16 + e)
//   n = (lane&15) + 16*tile ;  k = kchunk*32 + (lane>>4)*16 + e
// ---------------------------------------------------------------------------
__global__ void kmeans_prep(const float* __restrict__ cent,
                            float* __restrict__ cnorm,
                            __bf16* __restrict__ bpack, int K) {
    int gid = blockIdx.x * blockDim.x + threadIdx.x;
    if (gid < K * DIM) {
        int e    = gid & 15;
        int lane = (gid >> 4) & 31;
        int t    = (gid >> 9) & 15;
        int kch  = gid >> 13;
        int n = (lane & 15) + 16 * t;
        int k = kch * 32 + (lane >> 4) * 16 + e;
        bpack[gid] = (__bf16)cent[n * DIM + k];
    }
    if (gid < K) {
        float s = 0.0f;
        for (int d = 0; d < DIM; ++d) {
            float v = cent[gid * DIM + d];
            s += v * v;
        }
        cnorm[gid] = s;
    }
}

// ---------------------------------------------------------------------------
// Main kernel: WMMA scoring + in-register argmin + scatter accumulation.
// Block = 256 threads = 8 waves; each wave owns 16 rows.
// B matrix (64 KB bf16) is staged in LDS once per block; fragments are
// ping-pong register buffered (unroll 2) so ds_loads for tile t+1 overlap
// WMMAs of tile t with no inter-buffer register copies.
// ---------------------------------------------------------------------------
__global__ __launch_bounds__(256) void kmeans_assign(
    const float*  __restrict__ keys,
    const float*  __restrict__ values,
    const __bf16* __restrict__ bpack,
    const float*  __restrict__ cnorm,
    float* __restrict__ out_idx,     // [N] cluster index (as float)
    float* __restrict__ counts,      // [K]
    float* __restrict__ keySums,     // [K, D]
    float* __restrict__ valSums,     // [K, D]
    int N)
{
    __shared__ __align__(16) __bf16 s_b[BPACK_N];   // 64 KB

    // cooperative stage of the packed B matrix (4096 x 16B)
    {
        const uint4* src = (const uint4*)bpack;
        uint4*       dst = (uint4*)s_b;
        #pragma unroll
        for (int i = 0; i < (BPACK_N * 2 / 16) / 256; ++i)
            dst[threadIdx.x + i * 256] = src[threadIdx.x + i * 256];
    }
    __syncthreads();

    const int tid  = threadIdx.x;
    const int lane = tid & 31;
    const int wave = tid >> 5;
    const int rowBase = (blockIdx.x * 8 + wave) * 16;
    if (rowBase >= N) return;          // uniform per wave: EXEC stays all-1s

    const int m    = lane & 15;        // row within tile (A) / col within tile (C)
    const int half = lane >> 4;
    const int klo  = half * 16;

    // --- A fragments: 16x32 bf16 per k-chunk, contiguous k order (matches B pack) ---
    // lane<16 : M=lane,    K = 0..15
    // lane>=16: M=lane-16, K = 16..31
    v16bf afrag[KCHUNKS];
    const float* krow = keys + (size_t)(rowBase + m) * DIM;
    #pragma unroll
    for (int kc = 0; kc < KCHUNKS; ++kc) {
        const float4 f0 = *(const float4*)(krow + kc * 32 + klo);
        const float4 f1 = *(const float4*)(krow + kc * 32 + klo + 4);
        const float4 f2 = *(const float4*)(krow + kc * 32 + klo + 8);
        const float4 f3 = *(const float4*)(krow + kc * 32 + klo + 12);
        afrag[kc][0]  = (__bf16)f0.x; afrag[kc][1]  = (__bf16)f0.y;
        afrag[kc][2]  = (__bf16)f0.z; afrag[kc][3]  = (__bf16)f0.w;
        afrag[kc][4]  = (__bf16)f1.x; afrag[kc][5]  = (__bf16)f1.y;
        afrag[kc][6]  = (__bf16)f1.z; afrag[kc][7]  = (__bf16)f1.w;
        afrag[kc][8]  = (__bf16)f2.x; afrag[kc][9]  = (__bf16)f2.y;
        afrag[kc][10] = (__bf16)f2.z; afrag[kc][11] = (__bf16)f2.w;
        afrag[kc][12] = (__bf16)f3.x; afrag[kc][13] = (__bf16)f3.y;
        afrag[kc][14] = (__bf16)f3.z; afrag[kc][15] = (__bf16)f3.w;
    }

    // preload the 16 per-lane centroid norms (no in-loop loads/waits)
    float cn[NTILES];
    #pragma unroll
    for (int t = 0; t < NTILES; ++t) cn[t] = cnorm[m + 16 * t];

    // --- running argmin of score = ||c||^2 - 2*(x.c) per C element ---
    float best[8];
    int   bidx[8];
    #pragma unroll
    for (int j = 0; j < 8; ++j) { best[j] = 3.4e38f; bidx[j] = 0; }

    // ping-pong B fragment buffers out of LDS
    v16bf bbuf[2][KCHUNKS];
    #pragma unroll
    for (int kc = 0; kc < KCHUNKS; ++kc)
        bbuf[0][kc] = *(const v16bf*)(s_b + ((kc * NTILES) * 32 + lane) * 16);

    #pragma unroll 2
    for (int t = 0; t < NTILES; ++t) {
        const int cur = t & 1;          // constant per unrolled body
        const int nxt = cur ^ 1;
        if (t + 1 < NTILES) {           // prefetch next tile's fragments
            #pragma unroll
            for (int kc = 0; kc < KCHUNKS; ++kc)
                bbuf[nxt][kc] = *(const v16bf*)(
                    s_b + ((kc * NTILES + t + 1) * 32 + lane) * 16);
        }
        v8f acc = {};
        #pragma unroll
        for (int kc = 0; kc < KCHUNKS; ++kc)
            acc = __builtin_amdgcn_wmma_f32_16x16x32_bf16(
                false, afrag[kc], false, bbuf[cur][kc], (short)0, acc,
                false, false);

        const int n = m + 16 * t;              // C layout: N = lane&15 (+tile)
        #pragma unroll
        for (int j = 0; j < 8; ++j) {
            float score = __builtin_fmaf(-2.0f, acc[j], cn[t]);  // row M=j+8*half
            if (score < best[j]) { best[j] = score; bidx[j] = n; }
        }
    }

    // --- butterfly min-reduce over the 16 lanes of each half (wave32) ---
    #pragma unroll
    for (int off = 8; off >= 1; off >>= 1) {
        #pragma unroll
        for (int j = 0; j < 8; ++j) {
            float os = __shfl_xor(best[j], off, 32);
            int   oi = __shfl_xor(bidx[j], off, 32);
            if (os < best[j] || (os == best[j] && oi < bidx[j])) {
                best[j] = os; bidx[j] = oi;
            }
        }
    }

    // lanes 0 and 16 hold results for rows 0-7 / 8-15 respectively
    if (m == 0) {
        #pragma unroll
        for (int j = 0; j < 8; ++j)
            out_idx[rowBase + half * 8 + j] = (float)bidx[j];
    }

    // --- scatter-accumulate counts / key sums / value sums ---
    #pragma unroll
    for (int r = 0; r < 16; ++r) {
        const int src = (r < 8) ? 0 : 16;
        const int ci  = __shfl(bidx[r & 7], src, 32);
        const float* kr = keys   + (size_t)(rowBase + r) * DIM;
        const float* vr = values + (size_t)(rowBase + r) * DIM;
        float* ks = keySums + (size_t)ci * DIM;
        float* vs = valSums + (size_t)ci * DIM;
        #pragma unroll
        for (int q = 0; q < 4; ++q) {
            atomicAdd(&ks[lane * 4 + q], kr[lane * 4 + q]);
            atomicAdd(&vs[lane * 4 + q], vr[lane * 4 + q]);
        }
        if (lane == 0) atomicAdd(&counts[ci], 1.0f);
    }
}

// ---------------------------------------------------------------------------
// EMA update. One block per cluster, 128 threads (one per dim).
// ---------------------------------------------------------------------------
__global__ void kmeans_finalize(const float* __restrict__ kcent,
                                const float* __restrict__ vcent,
                                const float* __restrict__ ccounts,
                                const float* __restrict__ counts,
                                const float* __restrict__ keySums,
                                const float* __restrict__ valSums,
                                float* __restrict__ outKC,
                                float* __restrict__ outVC,
                                float* __restrict__ outCnt) {
    const int k = blockIdx.x;
    const int d = threadIdx.x;
    const float cnt  = counts[k];
    const float safe = fmaxf(cnt, 1.0f);
    const bool  ne   = cnt > 0.0f;
    const size_t i = (size_t)k * DIM + d;
    const float km = keySums[i] / safe;
    const float vm = valSums[i] / safe;
    const float okc = kcent[i], ovc = vcent[i];
    outKC[i] = ne ? (0.99f * okc + 0.01f * km) : okc;
    outVC[i] = ne ? (0.99f * ovc + 0.01f * vm) : ovc;
    if (d == 0) outCnt[k] = ccounts[k] + cnt;
}

// ---------------------------------------------------------------------------
extern "C" void kernel_launch(void* const* d_in, const int* in_sizes, int n_in,
                              void* d_out, int out_size, void* d_ws, size_t ws_size,
                              hipStream_t stream) {
    const float* keys            = (const float*)d_in[0];
    const float* values          = (const float*)d_in[1];
    const float* key_centroids   = (const float*)d_in[2];
    const float* value_centroids = (const float*)d_in[3];
    const float* cluster_counts  = (const float*)d_in[4];

    const int N = in_sizes[0] / DIM;   // 262144 rows
    const int K = in_sizes[2] / DIM;   // 256 clusters

    // workspace layout (floats, then bf16 pack; all >=32B aligned)
    float* ws      = (float*)d_ws;
    float* counts  = ws;                    // K
    float* keySums = counts + K;            // K*DIM
    float* valSums = keySums + (size_t)K * DIM;
    float* cnorm   = valSums + (size_t)K * DIM;   // K
    __bf16* bpack  = (__bf16*)(cnorm + K);        // K*DIM bf16

    // output layout (reference tuple, concatenated flat)
    float* out     = (float*)d_out;
    float* out_idx = out;                         // N
    float* outKC   = out_idx + N;                 // K*DIM
    float* outVC   = outKC + (size_t)K * DIM;     // K*DIM
    float* outCnt  = outVC + (size_t)K * DIM;     // K

    const int zn = K + 2 * K * DIM;
    kmeans_zero<<<(zn + 255) / 256, 256, 0, stream>>>(ws, zn);
    kmeans_prep<<<(K * DIM + 255) / 256, 256, 0, stream>>>(key_centroids, cnorm, bpack, K);
    kmeans_assign<<<(N + 127) / 128, 256, 0, stream>>>(
        keys, values, bpack, cnorm, out_idx, counts, keySums, valSums, N);
    kmeans_finalize<<<K, DIM, 0, stream>>>(
        key_centroids, value_centroids, cluster_counts,
        counts, keySums, valSums, outKC, outVC, outCnt);
}
